// GPT_11871289606298
// MI455X (gfx1250) — matrix operand order repaired and checked by reading, not compiled, not run on hardware
//
#include <hip/hip_runtime.h>
#include <cmath>

// ---------------------------------------------------------------------------
// GPT forward on gfx1250 (MI455X). All matmuls via v_wmma_f32_16x16x32_f16.
// f32 weights converted to f16 (v_cvt_pk_rtz) while filling LDS; weights are
// read once from HBM, B-tile re-reads across the 8 M-blocks hit the 192MB L2.
// Global->register->LDS single-buffer pipeline hides load latency under WMMA.
// ---------------------------------------------------------------------------

typedef _Float16 v16h __attribute__((ext_vector_type(16)));
typedef __fp16   v2fp16 __attribute__((ext_vector_type(2)));
typedef float    v8f  __attribute__((ext_vector_type(8)));

#define BM 128
#define BN 128
#define BK 32
#define LDS_STRIDE 40   // 32 + 8 halfs pad to spread LDS banks

// pack two f32 -> packed f16x2 dword (v_cvt_pk_rtz_f16_f32)
__device__ inline unsigned pack2(float lo, float hi) {
  v2fp16 p = __builtin_amdgcn_cvt_pkrtz(lo, hi);
  unsigned u; __builtin_memcpy(&u, &p, 4); return u;
}

// ---- WMMA fragment loaders (ISA 16-bit layouts, wave32) -------------------
// A 16x32 f16: lane L(0..15)=row M=L; VGPR v holds K = (v>=4?16:0) +
// (L>=16?8:0) + (v&3)*2 .. +1   -> two contiguous 16B LDS reads per fragment
__device__ inline v16h frag_a(const _Float16 T[][LDS_STRIDE], int rbase, int lane) {
  v16h f;
  const int row = rbase + (lane & 15);
  const int h8  = (lane >> 4) << 3;
#pragma unroll
  for (int v = 0; v < 8; ++v) {
    const int kb = ((v >> 2) << 4) + h8 + ((v & 3) << 1);
    f[2 * v]     = T[row][kb];
    f[2 * v + 1] = T[row][kb + 1];
  }
  return f;
}

// B 32x16 f16 (stored col-major in LDS as T[col][k]):
// lane holds N = lane&15, VGPR v -> K = (v>=4?16:0) + (lane>=16?8:0) + (v&3)*2 .. +1
__device__ inline v16h frag_b(const _Float16 T[][LDS_STRIDE], int cbase, int lane) {
  v16h f;
  const int col = cbase + (lane & 15);
  const int ko  = (lane >> 4) << 3;
#pragma unroll
  for (int v = 0; v < 8; ++v) {
    const int kb = ((v >> 2) << 4) + ko + ((v & 3) << 1);
    f[2 * v]     = T[col][kb];
    f[2 * v + 1] = T[col][kb + 1];
  }
  return f;
}

// ---------------------------------------------------------------------------
// Generic WMMA GEMM: C = op(A(f16) @ B + bias [,relu] [,+resid]).
// Block: 256 threads = 8 waves; tile 128x128x32; each wave = 64x32 = 4x2 WMMAs.
// M, N multiples of 128; K multiple of 32 (true for every call site).
// grid.z selects a slice via aZ/bZ/cZ element strides (used per-head).
// ---------------------------------------------------------------------------
__global__ __launch_bounds__(256) void gemm_wmma_f16(
    const _Float16* __restrict__ A, int lda, long long aZ,
    const void* __restrict__ B, int ldb, long long bZ, int bF16,
    const float* __restrict__ bias,
    const float* __restrict__ resid,
    float* __restrict__ Cf, _Float16* __restrict__ Ch,
    int ldc, long long cZ, int K, int relu)
{
  __shared__ __align__(16) _Float16 As[BM][LDS_STRIDE];
  __shared__ __align__(16) _Float16 Bs[BN][LDS_STRIDE];

  const int t    = threadIdx.x;
  const int lane = t & 31;
  const int w    = t >> 5, wm = w >> 2, wn = w & 3;
  const int m0   = blockIdx.y * BM;
  const int n0   = blockIdx.x * BN;
  const long long aoff = (long long)blockIdx.z * aZ;
  const long long boff = (long long)blockIdx.z * bZ;
  const long long coff = (long long)blockIdx.z * cZ;

  // A staging: thread owns 1 row x 16 halfs
  const int ar = t >> 1, ac0 = (t & 1) << 4;
  const _Float16* aSrc = A + aoff + (long long)(m0 + ar) * lda + ac0;
  // B staging: thread owns a k-pair x 8 cols (half-wave reads contiguous 512B)
  const int bk2 = (t >> 4) << 1, bc0 = (t & 15) << 3;
  const float*    bSrcF = (const float*)B    + boff + (long long)bk2 * ldb + n0 + bc0;
  const _Float16* bSrcH = (const _Float16*)B + boff + (long long)bk2 * ldb + n0 + bc0;

  uint4  aR0, aR1;
  float4 bf00, bf01, bf10, bf11;   // f32 weight path (rows k, k+1)
  uint4  bhR0, bhR1;               // f16 path (rows k, k+1)

  auto loadTiles = [&](int k0) {
    const _Float16* as = aSrc + k0;
    aR0 = *(const uint4*)(as);
    aR1 = *(const uint4*)(as + 8);
    if (bF16) {
      const _Float16* bs = bSrcH + (long long)k0 * ldb;
      bhR0 = *(const uint4*)(bs);
      bhR1 = *(const uint4*)(bs + ldb);
    } else {
      const float* bs = bSrcF + (long long)k0 * ldb;
      bf00 = *(const float4*)(bs);
      bf01 = *(const float4*)(bs + 4);
      bf10 = *(const float4*)(bs + ldb);
      bf11 = *(const float4*)(bs + ldb + 4);
      // warm L2 for the HBM-resident weight stream (2 K-tiles ahead,
      // speculative prefetch: OOB translations are silently dropped)
      __builtin_prefetch(bs + (long long)2 * BK * ldb, 0, 1);
    }
  };
  auto storeTiles = [&]() {
    *(uint4*)&As[ar][ac0]     = aR0;
    *(uint4*)&As[ar][ac0 + 8] = aR1;
    if (bF16) {
      const unsigned short* lo = (const unsigned short*)&bhR0;
      const unsigned short* hi = (const unsigned short*)&bhR1;
#pragma unroll
      for (int j = 0; j < 8; ++j)
        *(unsigned*)&Bs[bc0 + j][bk2] = (unsigned)lo[j] | ((unsigned)hi[j] << 16);
    } else {
      *(unsigned*)&Bs[bc0 + 0][bk2] = pack2(bf00.x, bf10.x);
      *(unsigned*)&Bs[bc0 + 1][bk2] = pack2(bf00.y, bf10.y);
      *(unsigned*)&Bs[bc0 + 2][bk2] = pack2(bf00.z, bf10.z);
      *(unsigned*)&Bs[bc0 + 3][bk2] = pack2(bf00.w, bf10.w);
      *(unsigned*)&Bs[bc0 + 4][bk2] = pack2(bf01.x, bf11.x);
      *(unsigned*)&Bs[bc0 + 5][bk2] = pack2(bf01.y, bf11.y);
      *(unsigned*)&Bs[bc0 + 6][bk2] = pack2(bf01.z, bf11.z);
      *(unsigned*)&Bs[bc0 + 7][bk2] = pack2(bf01.w, bf11.w);
    }
  };

  v8f acc[4][2];
#pragma unroll
  for (int i = 0; i < 4; ++i)
#pragma unroll
    for (int j = 0; j < 2; ++j)
#pragma unroll
      for (int e = 0; e < 8; ++e) acc[i][j][e] = 0.0f;

  loadTiles(0);
  for (int k0 = 0; k0 < K; k0 += BK) {
    __syncthreads();
    storeTiles();
    __syncthreads();
    if (k0 + BK < K) loadTiles(k0 + BK);  // overlap next loads with WMMA below

    v16h af[4], bfr[2];
#pragma unroll
    for (int i = 0; i < 4; ++i) af[i]  = frag_a(As, wm * 64 + i * 16, lane);
#pragma unroll
    for (int j = 0; j < 2; ++j) bfr[j] = frag_b(Bs, wn * 32 + j * 16, lane);
#pragma unroll
    for (int i = 0; i < 4; ++i)
#pragma unroll
      for (int j = 0; j < 2; ++j)
        acc[i][j] = __builtin_amdgcn_wmma_f32_16x16x32_f16(
            false, af[i], false, bfr[j], (short)0, acc[i][j], false, false);
  }

  // Epilogue: bias -> relu -> +residual; optional f32 and f16 outputs
  const int rsel = (lane >> 4) << 3;
  const int cw   = lane & 15;
#pragma unroll
  for (int i = 0; i < 4; ++i)
#pragma unroll
    for (int j = 0; j < 2; ++j) {
      const int gc = n0 + wn * 32 + j * 16 + cw;
      const float bv = bias ? bias[gc] : 0.0f;
#pragma unroll
      for (int v = 0; v < 8; ++v) {
        const int gr = m0 + wm * 64 + i * 16 + rsel + v;
        float val = acc[i][j][v] + bv;
        if (relu) val = fmaxf(val, 0.0f);
        if (resid) val += resid[(long long)gr * ldc + gc];
        const long long o = coff + (long long)gr * ldc + gc;
        if (Cf) Cf[o] = val;
        if (Ch) Ch[o] = (_Float16)val;
      }
    }
}

// ---------------------------------------------------------------------------
// scores[h,q,k] = (Q_h[q,:] . K_h[k,:]) / sqrt(d_model), masked (-inf, k>q).
// Same WMMA tile engine; key rows are contiguous per LDS column -> 16B stores.
// ---------------------------------------------------------------------------
__global__ __launch_bounds__(256) void qk_scores_wmma(
    const _Float16* __restrict__ qkvh, float* __restrict__ scores, float scale)
{
  __shared__ __align__(16) _Float16 As[BM][LDS_STRIDE];
  __shared__ __align__(16) _Float16 Bs[BN][LDS_STRIDE];

  const int t = threadIdx.x, lane = t & 31;
  const int w = t >> 5, wm = w >> 2, wn = w & 3;
  const int m0 = blockIdx.y * BM, n0 = blockIdx.x * BN, h = blockIdx.z;
  const _Float16* Q  = qkvh + h * 128;          // [q][d], stride 4608
  const _Float16* Kp = qkvh + 1536 + h * 128;   // [k][d], stride 4608

  const int ar = t >> 1, ac0 = (t & 1) << 4;
  const _Float16* aSrc = Q  + (long long)(m0 + ar) * 4608 + ac0;
  const _Float16* bSrc = Kp + (long long)(n0 + ar) * 4608 + ac0;

  uint4 aR0, aR1, bR0, bR1;
  auto loadTiles = [&](int k0) {
    aR0 = *(const uint4*)(aSrc + k0);
    aR1 = *(const uint4*)(aSrc + k0 + 8);
    bR0 = *(const uint4*)(bSrc + k0);
    bR1 = *(const uint4*)(bSrc + k0 + 8);
  };
  auto storeTiles = [&]() {
    *(uint4*)&As[ar][ac0]     = aR0;
    *(uint4*)&As[ar][ac0 + 8] = aR1;
    *(uint4*)&Bs[ar][ac0]     = bR0;
    *(uint4*)&Bs[ar][ac0 + 8] = bR1;
  };

  v8f acc[4][2];
#pragma unroll
  for (int i = 0; i < 4; ++i)
#pragma unroll
    for (int j = 0; j < 2; ++j)
#pragma unroll
      for (int e = 0; e < 8; ++e) acc[i][j][e] = 0.0f;

  loadTiles(0);
  for (int k0 = 0; k0 < 128; k0 += BK) {
    __syncthreads();
    storeTiles();
    __syncthreads();
    if (k0 + BK < 128) loadTiles(k0 + BK);

    v16h af[4], bfr[2];
#pragma unroll
    for (int i = 0; i < 4; ++i) af[i]  = frag_a(As, wm * 64 + i * 16, lane);
#pragma unroll
    for (int j = 0; j < 2; ++j) bfr[j] = frag_b(Bs, wn * 32 + j * 16, lane);
#pragma unroll
    for (int i = 0; i < 4; ++i)
#pragma unroll
      for (int j = 0; j < 2; ++j)
        acc[i][j] = __builtin_amdgcn_wmma_f32_16x16x32_f16(
            false, af[i], false, bfr[j], (short)0, acc[i][j], false, false);
  }

  float* out = scores + (long long)h * 1024 * 1024;
  const int rsel = (lane >> 4) << 3;
  const int cw   = lane & 15;
#pragma unroll
  for (int i = 0; i < 4; ++i)
#pragma unroll
    for (int j = 0; j < 2; ++j) {
      const int kk = n0 + wn * 32 + j * 16 + cw;
#pragma unroll
      for (int v = 0; v < 8; ++v) {
        const int q = m0 + wm * 64 + i * 16 + rsel + v;
        float val = acc[i][j][v] * scale;
        if (q < kk) val = -INFINITY;   // causal mask (mask==0 -> -inf)
        out[(long long)q * 1024 + kk] = val;
      }
    }
}

// ---------------------------------------------------------------------------
// Softmax over the QUERY axis (reference quirk: axis=-2 of (b,h,q,k)).
// One thread per key column; coalesced row sweeps. Writes f16 probabilities.
// ---------------------------------------------------------------------------
__global__ __launch_bounds__(256) void softmax_q(
    const float* __restrict__ scores, _Float16* __restrict__ attnh)
{
  const int c = blockIdx.x * blockDim.x + threadIdx.x;
  const int h = blockIdx.y;
  const float* sc = scores + (long long)h * 1024 * 1024 + c;
  _Float16*    at = attnh  + (long long)h * 1024 * 1024 + c;

  float m = -INFINITY;
  for (int q = 0; q < 1024; ++q) m = fmaxf(m, sc[(long long)q * 1024]);
  float s = 0.0f;
  for (int q = 0; q < 1024; ++q) {
    const float e = expf(sc[(long long)q * 1024] - m);  // -inf -> 0
    s += e;
    at[(long long)q * 1024] = (_Float16)e;              // e <= 1, safe in f16
  }
  const float inv = 1.0f / s;
  for (int q = 0; q < 1024; ++q)
    at[(long long)q * 1024] = (_Float16)((float)at[(long long)q * 1024] * inv);
}

// x[s] = tok_emb[idx[s]] + pos_emb[0]   (faithful quirk: pos_emb[:B], B==1)
__global__ __launch_bounds__(256) void embed_kernel(
    const int* __restrict__ idx, const float* __restrict__ tok,
    const float* __restrict__ pos, float* __restrict__ x,
    _Float16* __restrict__ xh)
{
  const int s = blockIdx.x;
  const float* te = tok + (long long)idx[s] * 1536;
  for (int d = threadIdx.x; d < 1536; d += blockDim.x) {
    const float v = te[d] + pos[d];
    x [(long long)s * 1536 + d] = v;
    xh[(long long)s * 1536 + d] = (_Float16)v;
  }
}

// Final LayerNorm (eps=1e-5), writes f16 for the output projection.
__global__ __launch_bounds__(256) void layernorm_kernel(
    const float* __restrict__ x, const float* __restrict__ g,
    const float* __restrict__ b, _Float16* __restrict__ xnh)
{
  __shared__ float red[256];
  const int s = blockIdx.x, t = threadIdx.x;
  const float* row = x + (long long)s * 1536;
  float sum = 0.0f, sq = 0.0f;
  for (int d = t; d < 1536; d += 256) { const float v = row[d]; sum += v; sq += v * v; }
  red[t] = sum; __syncthreads();
  for (int o = 128; o > 0; o >>= 1) { if (t < o) red[t] += red[t + o]; __syncthreads(); }
  const float mu = red[0] / 1536.0f; __syncthreads();
  red[t] = sq; __syncthreads();
  for (int o = 128; o > 0; o >>= 1) { if (t < o) red[t] += red[t + o]; __syncthreads(); }
  const float var = red[0] / 1536.0f - mu * mu;
  const float inv = rsqrtf(var + 1e-5f);
  for (int d = t; d < 1536; d += 256) {
    const float v = (row[d] - mu) * inv * g[d] + b[d];
    xnh[(long long)s * 1536 + d] = (_Float16)v;
  }
}

// ---------------------------------------------------------------------------
extern "C" void kernel_launch(void* const* d_in, const int* in_sizes, int n_in,
                              void* d_out, int out_size, void* d_ws, size_t ws_size,
                              hipStream_t stream)
{
  (void)in_sizes; (void)n_in; (void)out_size; (void)ws_size;
  constexpr int D = 1536, H = 12, L = 12, S = 1024;
  constexpr int D3 = 3 * D, D4 = 4 * D, HD = D / H;   // HD = 128

  const int*   idx  = (const int*)  d_in[0];
  /* d_in[1] = mask: fixed causal tril, applied analytically */
  const float* tok  = (const float*)d_in[2];
  const float* pos  = (const float*)d_in[3];
  const float* Wqkv = (const float*)d_in[4];
  const float* bqkv = (const float*)d_in[5];
  const float* Wo   = (const float*)d_in[6];
  const float* bo   = (const float*)d_in[7];
  const float* W1   = (const float*)d_in[8];
  const float* b1   = (const float*)d_in[9];
  const float* W2   = (const float*)d_in[10];
  const float* b2   = (const float*)d_in[11];
  const float* lng  = (const float*)d_in[12];
  const float* lnb  = (const float*)d_in[13];
  const float* Wout = (const float*)d_in[14];
  const float* bout = (const float*)d_in[15];

  char* ws = (char*)d_ws; size_t off = 0;
  auto alloc = [&](size_t bytes) {
    void* p = ws + off; off = (off + bytes + 255) & ~(size_t)255; return p;
  };
  float*    x      = (float*)   alloc((size_t)S * D  * 4);   //  6 MB activations
  _Float16* xh     = (_Float16*)alloc((size_t)S * D  * 2);   //  3 MB f16 acts
  _Float16* qkvh   = (_Float16*)alloc((size_t)S * D3 * 2);   //  9 MB q|k|v f16
  float*    scores = (float*)   alloc((size_t)H * S * S * 4);// 48 MB
  _Float16* attnh  = (_Float16*)alloc((size_t)H * S * S * 2);// 24 MB
  _Float16* oh     = (_Float16*)alloc((size_t)S * D  * 2);   //  3 MB
  _Float16* hh     = (_Float16*)alloc((size_t)S * D4 * 2);   // 12 MB
  _Float16* xnh    = (_Float16*)alloc((size_t)S * D  * 2);   //  3 MB

  embed_kernel<<<S, 256, 0, stream>>>(idx, tok, pos, x, xh);

  const float scale = 1.0f / sqrtf((float)D);  // quirk: sqrt(d_model)

  for (int l = 0; l < L; ++l) {
    // QKV: qkvh = f16(xh @ Wqkv[l] + bqkv[l])
    gemm_wmma_f16<<<dim3(D3 / 128, S / 128, 1), 256, 0, stream>>>(
        xh, D, 0, Wqkv + (size_t)l * D * D3, D3, 0, 0,
        bqkv + (size_t)l * D3, nullptr, nullptr, qkvh, D3, 0, D, 0);

    // scores = mask(Q K^T / sqrt(D)) per head
    qk_scores_wmma<<<dim3(S / 128, S / 128, H), 256, 0, stream>>>(qkvh, scores, scale);

    // softmax over query axis -> f16 probabilities
    softmax_q<<<dim3(S / 256, H), 256, 0, stream>>>(scores, attnh);

    // o_h = attn_h @ V_h   (per-head GEMM via grid.z strides)
    gemm_wmma_f16<<<dim3(HD / 128, S / 128, H), 256, 0, stream>>>(
        attnh, S, (long long)S * S, qkvh + 2 * D, D3, HD, 1,
        nullptr, nullptr, nullptr, oh, D, HD, S, 0);

    // x = x + (o @ Wo[l] + bo[l]);  also refresh xh
    gemm_wmma_f16<<<dim3(D / 128, S / 128, 1), 256, 0, stream>>>(
        oh, D, 0, Wo + (size_t)l * D * D, D, 0, 0,
        bo + (size_t)l * D, x, x, xh, D, 0, D, 0);

    // hh = f16(relu(xh @ W1[l] + b1[l]))
    gemm_wmma_f16<<<dim3(D4 / 128, S / 128, 1), 256, 0, stream>>>(
        xh, D, 0, W1 + (size_t)l * D * D4, D4, 0, 0,
        b1 + (size_t)l * D4, nullptr, nullptr, hh, D4, 0, D, 1);

    // x = x + (hh @ W2[l] + b2[l]);  refresh xh for next layer
    gemm_wmma_f16<<<dim3(D / 128, S / 128, 1), 256, 0, stream>>>(
        hh, D4, 0, W2 + (size_t)l * D4 * D, D, 0, 0,
        b2 + (size_t)l * D, x, x, xh, D, 0, D4, 0);
  }

  layernorm_kernel<<<S, 256, 0, stream>>>(x, lng, lnb, xnh);

  // out = LN(x) @ Wout + bout  (f32 output)
  gemm_wmma_f16<<<dim3(D / 128, S / 128, 1), 256, 0, stream>>>(
      xnh, D, 0, Wout, D, 0, 0, bout, nullptr,
      (float*)d_out, nullptr, D, 0, D, 0);
}
